// VQ_71614284694105
// MI455X (gfx1250) — matrix operand order repaired and checked by reading, not compiled, not run on hardware
//
#include <hip/hip_runtime.h>

// ---------------- problem constants (from reference) ----------------
#define BB     8
#define NNQ    4096
#define DIM    256
#define KK     4096
#define MROWS  (BB * NNQ)       // 32768 query rows
#define NTILES (KK / 16)        // 256 k-tiles of 16 codes
#define LDS_STRIDE 264          // DIM + 8 bf16 pad -> 528B rows (TDM pads 16B per 512B row)
#define TILE_B (16 * LDS_STRIDE * 2)   // 8448 bytes per staged 16x256 bf16 tile

typedef __attribute__((ext_vector_type(16))) __bf16       v16bf;
typedef __attribute__((ext_vector_type(8)))  float        v8f;
typedef __attribute__((ext_vector_type(4)))  float        f4;
typedef __attribute__((ext_vector_type(4)))  unsigned int u4;
typedef __attribute__((ext_vector_type(4)))  unsigned int ui4;
typedef __attribute__((ext_vector_type(8)))  int          i8v;
typedef __attribute__((ext_vector_type(4)))  int          i4v;

// Load a 32-byte B fragment (16 bf16) from LDS as two b128s.
__device__ __forceinline__ v16bf ld_frag(const __bf16* p) {
    union { u4 u[2]; v16bf v; } t;
    const u4* q = (const u4*)p;
    t.u[0] = q[0];
    t.u[1] = q[1];
    return t.v;
}

// LDS byte offset for a __shared__ pointer (strip flat aperture via addrspace(3)).
typedef __attribute__((address_space(3))) const void* lds_cptr;
__device__ __forceinline__ unsigned lds_byte_off(const void* p) {
    return (unsigned)(unsigned long long)(lds_cptr)p;
}

// ---- Tensor Data Mover: DMA one 16x256 bf16 tile (row-major, contiguous rows)
// into LDS at lds_off, inserting 16B padding after every 512B row (stride 264).
__device__ __forceinline__ void tdm_load_tile(const void* gsrc, unsigned lds_off) {
    const unsigned long long ga = (unsigned long long)(__UINTPTR_TYPE__)gsrc;
    ui4 g0;
    g0[0] = 1u;                                   // count=1 valid descriptor, user mode
    g0[1] = lds_off;                              // D#.lds_addr
    g0[2] = (unsigned)ga;                         // D#.global_addr[31:0]
    g0[3] = (unsigned)((ga >> 32) & 0x01FFFFFFu)  // D#.global_addr[56:32]
          | (2u << 30);                           // D#.type = 2 ("image")
    i8v g1;
    g1[0] = (1 << 16)      // data_size = 1 -> 2-byte elements
          | (1 << 20)      // pad_enable
          | (6 << 22)      // pad_interval: 128 DWORDs (=512B = one row)
          | (3 << 25);     // pad_amount: 4 DWORDs (=16B)
    g1[1] = (DIM & 0xFFFF) << 16;                 // tensor_dim0[15:0]=256 (bits 63:48)
    g1[2] = ((16 & 0xFFFF) << 16);                // tensor_dim0 hi=0, tensor_dim1[15:0]=16
    g1[3] = (DIM << 16);                          // tensor_dim1 hi=0, tile_dim0=256
    g1[4] = 16;                                   // tile_dim1=16 codes, tile_dim2=0
    g1[5] = DIM;                                  // tensor_dim0_stride[31:0]=256 elems
    g1[6] = 0;                                    // stride hi / dim1_stride lo
    g1[7] = 0;
    const i4v z4 = {0, 0, 0, 0};
#if __clang_major__ >= 23
    const i8v z8 = {0, 0, 0, 0, 0, 0, 0, 0};
    __builtin_amdgcn_tensor_load_to_lds(g0, g1, z4, z4, z8, 0);
#else
    __builtin_amdgcn_tensor_load_to_lds(g0, g1, z4, z4, 0);
#endif
}

// ---------------- kernel 1: decompose embed into bf16 hi/lo + |e|^2 ----------------
__global__ __launch_bounds__(256) void vq_prep(const float* __restrict__ embed,
                                               __bf16* __restrict__ ehi,
                                               __bf16* __restrict__ elo,
                                               float* __restrict__ enorm) {
    const int k = blockIdx.x;      // 4096 codes
    const int d = threadIdx.x;     // 256 dims
    const float v = embed[(size_t)k * DIM + d];
    const __bf16 h = (__bf16)v;
    ehi[(size_t)k * DIM + d] = h;
    elo[(size_t)k * DIM + d] = (__bf16)(v - (float)h);

    __shared__ float sm[256];
    sm[d] = v * v;
    __syncthreads();
    for (int s = 128; s > 0; s >>= 1) {
        if (d < s) sm[d] += sm[d + s];
        __syncthreads();
    }
    if (d == 0) enorm[k] = sm[0];
}

// ---------------- kernel 2: WMMA distance + argmin ----------------
// One wave owns 16 rows of x; sweeps all 4096 codes in 16-code tiles.
// dist = |e|^2 - 2 x.e   (|x|^2 dropped: constant per row)
// x.e via 3-term bf16 split: hi*hi + hi*lo (acc0) + lo*hi (acc1), f32 accumulate.
// Embed tiles are DMA'd into LDS by the Tensor Data Mover (wave 0 issues,
// TENSORcnt-synchronized, double-buffered).
__global__ __launch_bounds__(256) void vq_argmin(const float*  __restrict__ x,
                                                 const __bf16* __restrict__ ehi,
                                                 const __bf16* __restrict__ elo,
                                                 const float*  __restrict__ enorm,
                                                 int*          __restrict__ ind) {
    __shared__ __bf16 lds[2][2][16][LDS_STRIDE];   // [buf][hi/lo][code][dim] ~33.8 KB

    const int tid  = threadIdx.x;
    const int lane = tid & 31;
    const int wave = tid >> 5;
    const int m0   = (blockIdx.x * 8 + wave) * 16;
    const unsigned lbase = lds_byte_off(&lds[0][0][0][0]);

    // ---- kick off DMA of k-tile 0 into buffer 0 before touching x
    if (wave == 0) {
        tdm_load_tile(ehi, lbase);
        tdm_load_tile(elo, lbase + TILE_B);
    }

    // ---- load this wave's 16x256 x-tile once; decompose into resident A frags.
    // A frags use contiguous K here (B matches): lanes 0-15 row M=lane hold K=0..15;
    // lanes 16-31 row M=lane-16 hold K=16..31.
    const int    arow = m0 + (lane & 15);
    const int    doff = (lane & 16) ? 16 : 0;
    const float* xrow = x + (size_t)arow * DIM;

    v16bf ahi[8], alo[8];
#pragma unroll
    for (int c = 0; c < 8; ++c) {
        const float* base = xrow + c * 32 + doff;
        f4 p0 = *(const f4*)(base + 0);
        f4 p1 = *(const f4*)(base + 4);
        f4 p2 = *(const f4*)(base + 8);
        f4 p3 = *(const f4*)(base + 12);
        float vals[16];
#pragma unroll
        for (int i = 0; i < 4; ++i) {
            vals[i] = p0[i]; vals[4 + i] = p1[i]; vals[8 + i] = p2[i]; vals[12 + i] = p3[i];
        }
#pragma unroll
        for (int e = 0; e < 16; ++e) {
            const __bf16 h = (__bf16)vals[e];
            ahi[c][e] = h;
            alo[c][e] = (__bf16)(vals[e] - (float)h);
        }
    }

    float minv[8];
    int   mini[8];
#pragma unroll
    for (int r = 0; r < 8; ++r) { minv[r] = 3.4e38f; mini[r] = 0; }

    for (int t = 0; t < NTILES; ++t) {
        const int buf = t & 1;

        if (wave == 0) {
            if (t + 1 < NTILES) {
                // DMA next tile into the other buffer, then wait for the
                // CURRENT tile (tensor ops complete in order: keep newest 2 in flight).
                const __bf16* nh = ehi + (size_t)(t + 1) * 16 * DIM;
                const __bf16* nl = elo + (size_t)(t + 1) * 16 * DIM;
                const unsigned dst = lbase + (unsigned)(buf ^ 1) * (2u * TILE_B);
                tdm_load_tile(nh, dst);
                tdm_load_tile(nl, dst + TILE_B);
                __builtin_amdgcn_s_wait_tensorcnt(2);
            } else {
                __builtin_amdgcn_s_wait_tensorcnt(0);
            }
        }
        __syncthreads();   // tile t visible to all waves

        const int k0    = t * 16;
        const int bcode = lane & 15;
        const int bofs  = (lane & 16) ? 16 : 0;   // B: lanes 0-15 K=0..15, 16-31 K=16..31

        v8f acc0 = {};
        v8f acc1 = {};
#pragma unroll
        for (int c = 0; c < 8; ++c) {
            const v16bf bhi = ld_frag(&lds[buf][0][bcode][c * 32 + bofs]);
            const v16bf blo = ld_frag(&lds[buf][1][bcode][c * 32 + bofs]);
            acc0 = __builtin_amdgcn_wmma_f32_16x16x32_bf16(false, ahi[c], false, bhi,
                                                           (short)0, acc0, false, false);
            acc1 = __builtin_amdgcn_wmma_f32_16x16x32_bf16(false, alo[c], false, bhi,
                                                           (short)0, acc1, false, false);
            acc0 = __builtin_amdgcn_wmma_f32_16x16x32_bf16(false, ahi[c], false, blo,
                                                           (short)0, acc0, false, false);
        }

        // C layout: VGPR r = row m0+r (lanes 0-15) / m0+8+r (lanes 16-31), col = k0+lane%16
        const float en = enorm[k0 + bcode];
        const int   ki = k0 + bcode;
#pragma unroll
        for (int r = 0; r < 8; ++r) {
            const float dist = en - 2.0f * (acc0[r] + acc1[r]);
            if (dist < minv[r]) { minv[r] = dist; mini[r] = ki; }
        }
        __syncthreads();   // everyone done with buf before TDM overwrites it
    }

    // min-reduce across the 16 lanes of each half-wave (xor masks stay in-half)
#pragma unroll
    for (int r = 0; r < 8; ++r) {
#pragma unroll
        for (int off = 8; off > 0; off >>= 1) {
            const float ov = __shfl_xor(minv[r], off, 32);
            const int   oi = __shfl_xor(mini[r], off, 32);
            if (ov < minv[r] || (ov == minv[r] && oi < mini[r])) { minv[r] = ov; mini[r] = oi; }
        }
    }
    if (lane == 0) {
#pragma unroll
        for (int r = 0; r < 8; ++r) ind[m0 + r] = mini[r];
    } else if (lane == 16) {
#pragma unroll
        for (int r = 0; r < 8; ++r) ind[m0 + 8 + r] = mini[r];
    }
}

// ---------------- kernel 3: gather qx, straight-through out, per-row loss partial ----------------
__global__ __launch_bounds__(256) void vq_gather(const float* __restrict__ x,
                                                 const float* __restrict__ embed,
                                                 const int*   __restrict__ ind,
                                                 float* __restrict__ out,
                                                 float* __restrict__ partial) {
    const int row  = blockIdx.x;
    const int d    = threadIdx.x;
    const int code = ind[row];
    const float q  = embed[(size_t)code * DIM + d];
    const float xv = x[(size_t)row * DIM + d];
    out[(size_t)row * DIM + d] = (q - xv) + xv;   // stop_grad(qx-x)+x == qx (forward)
    const float diff = q - xv;

    __shared__ float sm[256];
    sm[d] = diff * diff;
    __syncthreads();
    for (int s = 128; s > 0; s >>= 1) {
        if (d < s) sm[d] += sm[d + s];
        __syncthreads();
    }
    if (d == 0) partial[row] = sm[0];
}

// ---------------- kernel 4: deterministic loss reduction ----------------
__global__ __launch_bounds__(256) void vq_finalize(const float* __restrict__ partial,
                                                   float* __restrict__ loss_out) {
    __shared__ float sm[256];
    float s = 0.f;
    for (int i = threadIdx.x; i < MROWS; i += 256) s += partial[i];  // fixed order
    sm[threadIdx.x] = s;
    __syncthreads();
    for (int st = 128; st > 0; st >>= 1) {
        if (threadIdx.x < st) sm[threadIdx.x] += sm[threadIdx.x + st];
        __syncthreads();
    }
    if (threadIdx.x == 0) {
        const float l = sm[0] / (float)((size_t)MROWS * DIM);
        loss_out[0] = l;   // loss1
        loss_out[1] = l;   // loss2 (forward-identical)
    }
}

// ---------------- launch ----------------
extern "C" void kernel_launch(void* const* d_in, const int* in_sizes, int n_in,
                              void* d_out, int out_size, void* d_ws, size_t ws_size,
                              hipStream_t stream) {
    const float* x     = (const float*)d_in[0];   // [8, 4096, 256] f32
    const float* embed = (const float*)d_in[1];   // [4096, 256]    f32
    float* out = (float*)d_out;                   // [8*4096*256] + 2 losses

    // workspace layout (~4.3 MB)
    char* ws = (char*)d_ws;
    __bf16* ehi     = (__bf16*)(ws);                                   // 2 MB
    __bf16* elo     = (__bf16*)(ws + (size_t)2 * 1024 * 1024);         // 2 MB
    float*  enorm   = (float*) (ws + (size_t)4 * 1024 * 1024);         // 16 KB
    int*    ind     = (int*)   (ws + (size_t)4 * 1024 * 1024 + 16 * 1024);          // 128 KB
    float*  partial = (float*) (ws + (size_t)4 * 1024 * 1024 + 16 * 1024 + 128 * 1024); // 128 KB

    vq_prep    <<<KK,             256, 0, stream>>>(embed, ehi, elo, enorm);
    vq_argmin  <<<MROWS / (8*16), 256, 0, stream>>>(x, ehi, elo, enorm, ind);
    vq_gather  <<<MROWS,          256, 0, stream>>>(x, embed, ind, out, partial);
    vq_finalize<<<1,              256, 0, stream>>>(partial, out + (size_t)MROWS * DIM);
}
